// MultiheadSelfAttentionWithEdgeBias_34986803593873
// MI455X (gfx1250) — matrix operand order repaired and checked by reading, not compiled, not run on hardware
//
#include <hip/hip_runtime.h>
#include <hip/hip_bf16.h>

#define N_TOK 4096
#define DMODEL 512
#define NHEAD 8
#define DKH 64
#define NE 8

typedef __attribute__((ext_vector_type(16))) _Float16 v16h;
typedef __attribute__((ext_vector_type(8)))  _Float16 v8h;
typedef __attribute__((ext_vector_type(8)))  float    v8f;
typedef __attribute__((ext_vector_type(4)))  unsigned int u32x4;
typedef __attribute__((ext_vector_type(8)))  int      i32x8;
typedef __attribute__((ext_vector_type(4)))  int      i32x4;

// ---------------------------------------------------------------------------
// Fragment helpers (CDNA5 WMMA 16x16x32 f16 layouts, wave32)
//  A (16x32, MxK): lane m = lane%16 holds row m; element e -> K = 16*hi+e
//                  (hi = lane>=16)
//  B (32x16, KxN): lane n = lane%16 holds col n; element e -> K = 16*hi+e
//  C/D (16x16 f32): VGPR r -> row r+8*hi, col = lane%16
// ---------------------------------------------------------------------------
__device__ inline v16h frag16(const _Float16* p0, const _Float16* p1) {
  v8h lo = *(const v8h*)p0;
  v8h hi = *(const v8h*)p1;
  v16h r;
#pragma unroll
  for (int i = 0; i < 8; ++i) { r[i] = lo[i]; r[i + 8] = hi[i]; }
  return r;
}

__device__ inline v8f wmma_f16(v16h a, v16h b, v8f c) {
  return __builtin_amdgcn_wmma_f32_16x16x32_f16(false, a, false, b,
                                                (short)0, c, false, false);
}

// ---------------------------------------------------------------------------
// Tensor Data Mover: 2-D tile load (f16 elements) global -> LDS
// D# per CDNA5 ISA: group0 {count, lds_addr, global_addr,
// type=2}, group1 {data_size=2B, tensor dims, tile dims, dim0 stride}.
// ---------------------------------------------------------------------------
__device__ inline void tdm_load_tile_2d(const _Float16* gsrc, void* lds_dst,
                                        unsigned tile_w, unsigned tile_h,
                                        unsigned stride_elems) {
  unsigned long long ga = (unsigned long long)gsrc;
  unsigned lds  = (unsigned)__builtin_amdgcn_readfirstlane(
                      (int)(unsigned)(unsigned long long)lds_dst);
  unsigned galo = (unsigned)__builtin_amdgcn_readfirstlane((int)(unsigned)ga);
  unsigned gahi = (unsigned)__builtin_amdgcn_readfirstlane((int)(unsigned)(ga >> 32));

  u32x4 g0;
  g0[0] = 1u;                                  // count=1, user descriptor
  g0[1] = lds;                                 // lds_addr (bytes)
  g0[2] = galo;                                // global_addr[31:0]
  g0[3] = (gahi & 0x01ffffffu) | 0x80000000u;  // global_addr[56:32] | type=2

  const unsigned TD0 = 1u << 20;               // tensor dims: large (tiles are
  const unsigned TD1 = 1u << 20;               // always fully in-bounds)
  i32x8 g1;
  g1[0] = (int)(1u << 16);                     // data_size=1 -> 2 bytes
  g1[1] = (int)((TD0 & 0xffffu) << 16);        // tensor_dim0 lo16
  g1[2] = (int)((TD0 >> 16) | ((TD1 & 0xffffu) << 16));
  g1[3] = (int)((TD1 >> 16) | (tile_w << 16)); // tile_dim0 = contiguous elems
  g1[4] = (int)(tile_h & 0xffffu);             // tile_dim1 (tile_dim2 = 0)
  g1[5] = (int)stride_elems;                   // tensor_dim0_stride lo32
  g1[6] = 0;
  g1[7] = 0;

  i32x4 z4 = {0, 0, 0, 0};
#if defined(__clang_major__) && (__clang_major__ >= 23)
  i32x8 z8 = {0, 0, 0, 0, 0, 0, 0, 0};
  __builtin_amdgcn_tensor_load_to_lds(g0, g1, z4, z4, z8, 0);
#else
  __builtin_amdgcn_tensor_load_to_lds(g0, g1, z4, z4, 0);
#endif
}

// ---------------------------------------------------------------------------
// Kernel 0: f32 -> f16 conversion
// ---------------------------------------------------------------------------
__global__ void cvt_f32_to_f16(const float* __restrict__ src,
                               _Float16* __restrict__ dst, int n) {
  int i = blockIdx.x * blockDim.x + threadIdx.x;
  if (i < n) dst[i] = (_Float16)src[i];
}

// ---------------------------------------------------------------------------
// Kernel 1: edge bias  c[i,j] = sum_e edges[i,j,e] * W1[e]   (512 MB stream)
// ---------------------------------------------------------------------------
__global__ void edge_bias_kernel(const float* __restrict__ edges,
                                 const float* __restrict__ w1,
                                 _Float16* __restrict__ cb) {
  long long idx = (long long)blockIdx.x * blockDim.x + threadIdx.x;
  if (idx >= (long long)N_TOK * N_TOK) return;
  const float4* p = reinterpret_cast<const float4*>(edges + idx * NE);
  float4 a = p[0];
  float4 b = p[1];
  float acc = a.x * w1[0] + a.y * w1[1] + a.z * w1[2] + a.w * w1[3] +
              b.x * w1[4] + b.y * w1[5] + b.z * w1[6] + b.w * w1[7];
  cb[idx] = (_Float16)acc;
}

// ---------------------------------------------------------------------------
// Kernel 2: QKV projection.  y = x @ W^T  (one wave -> 16x64 output tile)
// z = 0: Q -> qh[N][D]; z = 1: K -> kh[N][D]; z = 2: V -> vt[h][d][j] (transposed)
// ---------------------------------------------------------------------------
__global__ __launch_bounds__(32) void qkv_proj_kernel(
    const _Float16* __restrict__ xh, const _Float16* __restrict__ Wqh,
    const _Float16* __restrict__ Wkh, const _Float16* __restrict__ Wvh,
    _Float16* __restrict__ qh, _Float16* __restrict__ kh,
    _Float16* __restrict__ vtg) {
  const int lane = threadIdx.x & 31;
  const int hi = lane >> 4, lm = lane & 15;
  const int i0 = blockIdx.x * 16;
  const int n0 = blockIdx.y * 64;
  const int z = blockIdx.z;
  const _Float16* W = (z == 0) ? Wqh : (z == 1) ? Wkh : Wvh;

  v8f acc[4] = {};
  const _Float16* arow = xh + (i0 + lm) * DMODEL;
  for (int kt = 0; kt < DMODEL / 32; ++kt) {
    const int kb = kt * 32;
    v16h a = frag16(arow + kb + 16 * hi, arow + kb + 16 * hi + 8);
#pragma unroll
    for (int s = 0; s < 4; ++s) {
      const _Float16* brow = W + (n0 + s * 16 + lm) * DMODEL + kb + 16 * hi;
      v16h b = frag16(brow, brow + 8);
      acc[s] = wmma_f16(a, b, acc[s]);
    }
  }
#pragma unroll
  for (int s = 0; s < 4; ++s) {
#pragma unroll
    for (int r = 0; r < 8; ++r) {
      int row = i0 + r + 8 * hi;
      int col = n0 + s * 16 + lm;
      _Float16 val = (_Float16)acc[s][r];
      if (z == 0)      qh[row * DMODEL + col] = val;
      else if (z == 1) kh[row * DMODEL + col] = val;
      else {
        int h = col >> 6, d = col & 63;
        vtg[(h * DKH + d) * N_TOK + row] = val;   // V transposed per head
      }
    }
  }
}

// ---------------------------------------------------------------------------
// Kernel 3: flash attention with edge bias.
// Block = 128 threads (4 waves); each wave owns 16 query rows of one head.
// K / V^T tiles (32 keys per step) are TDM-loaded into LDS, double-buffered.
// ---------------------------------------------------------------------------
__global__ __launch_bounds__(128) void attention_kernel(
    const _Float16* __restrict__ qh, const _Float16* __restrict__ kh,
    const _Float16* __restrict__ vtg, const _Float16* __restrict__ cb,
    _Float16* __restrict__ ao) {
  __shared__ _Float16 Kt[2][32 * 64];   // [j_local][d]
  __shared__ _Float16 Vt[2][64 * 32];   // [d][j_local]
  __shared__ _Float16 Pst[4][16 * 32];  // per-wave P staging

  const int lane = threadIdx.x & 31;
  const int wave = threadIdx.x >> 5;
  const int hi = lane >> 4, lm = lane & 15;
  const int h = blockIdx.y;
  const int i0 = blockIdx.x * 64 + wave * 16;

  // Q fragments: fixed for the whole j loop.
  const _Float16* qrow = qh + (i0 + lm) * DMODEL + h * DKH;
  v16h aq0 = frag16(qrow + 16 * hi, qrow + 16 * hi + 8);
  v16h aq1 = frag16(qrow + 32 + 16 * hi, qrow + 32 + 16 * hi + 8);

  v8f acc[4] = {};
  float mrun[8], lrun[8];
#pragma unroll
  for (int r = 0; r < 8; ++r) { mrun[r] = -1e30f; lrun[r] = 0.f; }

  const int NJ = N_TOK / 32;
  if (threadIdx.x < 32) {  // wave 0 drives the TDM
    tdm_load_tile_2d(kh + 0 * DMODEL + h * DKH, &Kt[0][0], 64, 32, DMODEL);
    tdm_load_tile_2d(vtg + (long long)h * DKH * N_TOK + 0, &Vt[0][0], 32, 64, N_TOK);
  }

  for (int jt = 0; jt < NJ; ++jt) {
    const int buf = jt & 1;
    if (threadIdx.x < 32) {
      if (jt + 1 < NJ) {  // prefetch next chunk into the other buffer
        const int jn = (jt + 1) * 32;
        tdm_load_tile_2d(kh + (long long)jn * DMODEL + h * DKH,
                         &Kt[buf ^ 1][0], 64, 32, DMODEL);
        tdm_load_tile_2d(vtg + (long long)h * DKH * N_TOK + jn,
                         &Vt[buf ^ 1][0], 32, 64, N_TOK);
        __builtin_amdgcn_s_wait_tensorcnt(2);  // current pair done (in-order)
      } else {
        __builtin_amdgcn_s_wait_tensorcnt(0);
      }
    }
    __syncthreads();

    const _Float16* Kl = &Kt[buf][0];
    const _Float16* Vl = &Vt[buf][0];
    const int j0 = jt * 32;

    // S = Q K^T for two 16-wide key sub-tiles, + bias, * 1/sqrt(dk)
    v8f st[2];
#pragma unroll
    for (int t = 0; t < 2; ++t) {
      v8f s = {};
      const _Float16* k0 = Kl + (t * 16 + lm) * 64 + 16 * hi;
      s = wmma_f16(aq0, frag16(k0, k0 + 8), s);
      const _Float16* k1 = k0 + 32;
      s = wmma_f16(aq1, frag16(k1, k1 + 8), s);
#pragma unroll
      for (int r = 0; r < 8; ++r) {
        float c = (float)cb[(long long)(i0 + r + 8 * hi) * N_TOK +
                            (j0 + t * 16 + lm)];
        s[r] = s[r] * 0.125f + c;
      }
      st[t] = s;
    }

    // Online softmax (row reductions across 16 lanes of each half-wave).
#pragma unroll
    for (int r = 0; r < 8; ++r) {
      float mx = fmaxf(st[0][r], st[1][r]);
#pragma unroll
      for (int off = 8; off; off >>= 1) mx = fmaxf(mx, __shfl_xor(mx, off, 32));
      float mnew = fmaxf(mrun[r], mx);
      float rescale = __expf(mrun[r] - mnew);
      mrun[r] = mnew;
      float p0 = __expf(st[0][r] - mnew);
      float p1 = __expf(st[1][r] - mnew);
      float rs = p0 + p1;
#pragma unroll
      for (int off = 8; off; off >>= 1) rs += __shfl_xor(rs, off, 32);
      lrun[r] = lrun[r] * rescale + rs;
#pragma unroll
      for (int c = 0; c < 4; ++c) acc[c][r] *= rescale;
      const int prow = r + 8 * hi;
      Pst[wave][prow * 32 + lm]      = (_Float16)p0;
      Pst[wave][prow * 32 + 16 + lm] = (_Float16)p1;
    }

    // O += P V  (P re-read from LDS in A-fragment layout; LDS ops in-order
    // within a wave, so no barrier needed for this per-wave staging buffer).
    const _Float16* pp = &Pst[wave][lm * 32 + 16 * hi];
    v16h pa = frag16(pp, pp + 8);
#pragma unroll
    for (int c = 0; c < 4; ++c) {
      const _Float16* vv = Vl + (c * 16 + lm) * 32 + 16 * hi;
      acc[c] = wmma_f16(pa, frag16(vv, vv + 8), acc[c]);
    }
    __syncthreads();  // all reads of this buffer done before TDM reuses it
  }

  // Epilogue: normalize and store attention output (f16, [N][D]).
#pragma unroll
  for (int r = 0; r < 8; ++r) {
    float inv = 1.0f / lrun[r];
    int row = i0 + r + 8 * hi;
#pragma unroll
    for (int c = 0; c < 4; ++c)
      ao[row * DMODEL + h * DKH + c * 16 + lm] = (_Float16)(acc[c][r] * inv);
  }
}

// ---------------------------------------------------------------------------
// Kernel 4: output projection  out = ao @ Wo^T  (f32 result)
// ---------------------------------------------------------------------------
__global__ __launch_bounds__(32) void out_proj_kernel(
    const _Float16* __restrict__ ao, const _Float16* __restrict__ Woh,
    float* __restrict__ out) {
  const int lane = threadIdx.x & 31;
  const int hi = lane >> 4, lm = lane & 15;
  const int i0 = blockIdx.x * 16;
  const int n0 = blockIdx.y * 64;

  v8f acc[4] = {};
  const _Float16* arow = ao + (i0 + lm) * DMODEL;
  for (int kt = 0; kt < DMODEL / 32; ++kt) {
    const int kb = kt * 32;
    v16h a = frag16(arow + kb + 16 * hi, arow + kb + 16 * hi + 8);
#pragma unroll
    for (int s = 0; s < 4; ++s) {
      const _Float16* brow = Woh + (n0 + s * 16 + lm) * DMODEL + kb + 16 * hi;
      acc[s] = wmma_f16(a, frag16(brow, brow + 8), acc[s]);
    }
  }
#pragma unroll
  for (int s = 0; s < 4; ++s)
#pragma unroll
    for (int r = 0; r < 8; ++r)
      out[(i0 + r + 8 * hi) * DMODEL + n0 + s * 16 + lm] = acc[s][r];
}

// ---------------------------------------------------------------------------
extern "C" void kernel_launch(void* const* d_in, const int* in_sizes, int n_in,
                              void* d_out, int out_size, void* d_ws,
                              size_t ws_size, hipStream_t stream) {
  const float* x     = (const float*)d_in[0];
  const float* edges = (const float*)d_in[1];
  const float* Wq    = (const float*)d_in[2];
  const float* Wk    = (const float*)d_in[3];
  const float* Wv    = (const float*)d_in[4];
  const float* Wo    = (const float*)d_in[5];
  const float* W1    = (const float*)d_in[6];
  float* out = (float*)d_out;

  char* ws = (char*)d_ws;
  size_t off = 0;
  auto alloc = [&](size_t bytes) -> void* {
    off = (off + 255) & ~(size_t)255;
    void* p = ws + off;
    off += bytes;
    return p;
  };
  const size_t ND = (size_t)N_TOK * DMODEL, DD = (size_t)DMODEL * DMODEL;
  _Float16* xh  = (_Float16*)alloc(ND * 2);
  _Float16* wqh = (_Float16*)alloc(DD * 2);
  _Float16* wkh = (_Float16*)alloc(DD * 2);
  _Float16* wvh = (_Float16*)alloc(DD * 2);
  _Float16* woh = (_Float16*)alloc(DD * 2);
  _Float16* qh  = (_Float16*)alloc(ND * 2);
  _Float16* kh  = (_Float16*)alloc(ND * 2);
  _Float16* vtg = (_Float16*)alloc(ND * 2);
  _Float16* cb  = (_Float16*)alloc((size_t)N_TOK * N_TOK * 2);
  _Float16* ao  = (_Float16*)alloc(ND * 2);

  // 0) precision conversion
  cvt_f32_to_f16<<<(int)((ND + 255) / 256), 256, 0, stream>>>(x, xh, (int)ND);
  cvt_f32_to_f16<<<(int)((DD + 255) / 256), 256, 0, stream>>>(Wq, wqh, (int)DD);
  cvt_f32_to_f16<<<(int)((DD + 255) / 256), 256, 0, stream>>>(Wk, wkh, (int)DD);
  cvt_f32_to_f16<<<(int)((DD + 255) / 256), 256, 0, stream>>>(Wv, wvh, (int)DD);
  cvt_f32_to_f16<<<(int)((DD + 255) / 256), 256, 0, stream>>>(Wo, woh, (int)DD);

  // 1) edge bias: the 512 MB roofline stream
  edge_bias_kernel<<<(N_TOK * N_TOK) / 256, 256, 0, stream>>>(edges, W1, cb);

  // 2) QKV projections (WMMA)
  qkv_proj_kernel<<<dim3(N_TOK / 16, DMODEL / 64, 3), 32, 0, stream>>>(
      xh, wqh, wkh, wvh, qh, kh, vtg);

  // 3) flash attention with TDM-fed LDS tiles (WMMA)
  attention_kernel<<<dim3(N_TOK / 64, NHEAD), 128, 0, stream>>>(qh, kh, vtg,
                                                                cb, ao);

  // 4) output projection (WMMA) -> f32
  out_proj_kernel<<<dim3(N_TOK / 16, DMODEL / 64), 32, 0, stream>>>(ao, woh,
                                                                    out);
}
